// DrugGCN_85899345920324
// MI455X (gfx1250) — matrix operand order repaired and checked by reading, not compiled, not run on hardware
//
#include <hip/hip_runtime.h>
#include <hip/hip_bf16.h>

typedef __bf16 bf16;
typedef __attribute__((ext_vector_type(16))) __bf16 v16bf;
typedef __attribute__((ext_vector_type(8)))  __bf16 v8bf;
typedef __attribute__((ext_vector_type(8)))  float  v8f;

// ---------------------------------------------------------------- degrees
__global__ void k_deg_init(float* __restrict__ deg, int n) {
    int i = blockIdx.x * blockDim.x + threadIdx.x;
    if (i < n) deg[i] = 1.0f;                     // self-loop
}

__global__ void k_deg_edges(const long long* __restrict__ dst,
                            float* __restrict__ deg, int e) {
    int i = blockIdx.x * blockDim.x + threadIdx.x;
    if (i < e) unsafeAtomicAdd(&deg[dst[i]], 1.0f);
}

__global__ void k_dinv(const float* __restrict__ deg,
                       float* __restrict__ dinv, int n) {
    int i = blockIdx.x * blockDim.x + threadIdx.x;
    if (i < n) dinv[i] = rsqrtf(deg[i]);          // deg >= 1 always
}

// ---------------------------------------------------------------- f32 -> bf16
__global__ void k_f2bf(const float* __restrict__ in, bf16* __restrict__ out, int n) {
    int i = blockIdx.x * blockDim.x + threadIdx.x;
    if (i < n) out[i] = (bf16)in[i];
}

__global__ void k_relu_f2bf(const float* __restrict__ in, bf16* __restrict__ out, int n) {
    int i = blockIdx.x * blockDim.x + threadIdx.x;
    if (i < n) out[i] = (bf16)fmaxf(in[i], 0.0f);
}

// ---------------------------------------------------------------- pack W into WMMA B-fragment layout
// Wp[((nt*nkt + kt)*32 + lane)*16 + e] = W[(kt*32 + (lane<16?0:16) + e) * Fout + nt*16 + (lane&15)]
__global__ void k_pack_w(const float* __restrict__ W, bf16* __restrict__ Wp,
                         int K, int Fout) {
    int idx = blockIdx.x * blockDim.x + threadIdx.x;
    int total = K * Fout;
    if (idx >= total) return;
    int nkt  = K >> 5;
    int e    = idx & 15;
    int lane = (idx >> 4) & 31;
    int t    = idx >> 9;
    int kt   = t % nkt;
    int nt   = t / nkt;
    int k    = kt * 32 + ((lane < 16) ? 0 : 16) + e;
    int n    = nt * 16 + (lane & 15);
    Wp[idx]  = (bf16)W[(size_t)k * Fout + n];
}

// ---------------------------------------------------------------- bf16 WMMA GEMM: C[M,Fout] = A[M,128] @ W
// one wave per 16x16 tile; blockDim.x = 32 * (Fout/16); gridDim.x = M/16
__global__ void k_gemm_wmma(const bf16* __restrict__ A, const bf16* __restrict__ Bp,
                            float* __restrict__ C, int Fout) {
    const int K    = 128;
    const int nkt  = 4;                 // K / 32
    const int lane = threadIdx.x & 31;
    const int nt   = threadIdx.x >> 5;  // n-tile = wave id
    const int mt   = blockIdx.x;
    const int lm   = lane & 15;
    const int hi   = lane >> 4;         // 0: lanes 0-15, 1: lanes 16-31

    v8f acc = {};
    // A k order chosen here (contiguous, same as Wp): lane<16 holds K {0..15}; lane>=16 holds {16..31}
    const bf16* arow  = A + (size_t)(mt * 16 + lm) * K + (hi ? 16 : 0);
    const bf16* bbase = Bp + ((size_t)nt * nkt * 32 + lane) * 16;

#pragma unroll
    for (int kt = 0; kt < nkt; ++kt) {
        v8bf a0 = *(const v8bf*)(arow + kt * 32);
        v8bf a1 = *(const v8bf*)(arow + kt * 32 + 8);
        v16bf a = __builtin_shufflevector(a0, a1, 0,1,2,3,4,5,6,7,8,9,10,11,12,13,14,15);
        v16bf b = *(const v16bf*)(bbase + (size_t)kt * 32 * 16);
        acc = __builtin_amdgcn_wmma_f32_16x16x32_bf16(false, a, false, b,
                                                      (short)0, acc, false, false);
    }
    // C layout: lane<16 -> rows 0..7, lane>=16 -> rows 8..15; col = nt*16 + (lane&15)
    float* cp = C + (size_t)(mt * 16 + hi * 8) * Fout + nt * 16 + lm;
#pragma unroll
    for (int r = 0; r < 8; ++r)
        cp[(size_t)r * Fout] = acc[r];
}

// ---------------------------------------------------------------- self-loop term + bias
__global__ void k_selfloop(const float* __restrict__ h, const float* __restrict__ dinv,
                           const float* __restrict__ bias, float* __restrict__ out,
                           int fshift, int total) {
    int i = blockIdx.x * blockDim.x + threadIdx.x;
    if (i >= total) return;
    int node = i >> fshift;
    int f    = i & ((1 << fshift) - 1);
    float di = dinv[node];
    out[i] = h[i] * di * di + bias[f];
}

// ---------------------------------------------------------------- edge scatter-add
// blockDim = (F, EPB); grid = ceil(E/EPB)
__global__ void k_scatter(const float* __restrict__ h, const float* __restrict__ dinv,
                          const long long* __restrict__ src, const long long* __restrict__ dst,
                          float* __restrict__ out, int F, int E) {
    int e = blockIdx.x * blockDim.y + threadIdx.y;
    if (e >= E) return;
    long long s = src[e];
    long long d = dst[e];
    float norm = dinv[s] * dinv[d];
    int f = threadIdx.x;
    unsafeAtomicAdd(&out[(size_t)d * F + f], h[(size_t)s * F + f] * norm);
}

// ---------------------------------------------------------------- final relu + mean over nodes
__global__ void k_zero(float* __restrict__ out, int n) {
    int i = blockIdx.x * blockDim.x + threadIdx.x;
    if (i < n) out[i] = 0.0f;
}

__global__ void k_mean_relu(const float* __restrict__ x, float* __restrict__ out,
                            int N, int F, float invN) {
    int f = threadIdx.x;                     // F threads
    int chunk = (N + gridDim.x - 1) / gridDim.x;
    int r0 = blockIdx.x * chunk;
    int r1 = min(N, r0 + chunk);
    float s = 0.0f;
    for (int r = r0; r < r1; ++r)
        s += fmaxf(x[(size_t)r * F + f], 0.0f);
    unsafeAtomicAdd(&out[f], s * invN);
}

// ================================================================ launch
extern "C" void kernel_launch(void* const* d_in, const int* in_sizes, int n_in,
                              void* d_out, int out_size, void* d_ws, size_t ws_size,
                              hipStream_t stream) {
    const float*     x   = (const float*)d_in[0];
    const long long* ei  = (const long long*)d_in[1];   // int64 edge_index [2, E]
    const float*     W1  = (const float*)d_in[2];
    const float*     b1  = (const float*)d_in[3];
    const float*     W2  = (const float*)d_in[4];
    const float*     b2  = (const float*)d_in[5];
    float*           out = (float*)d_out;

    const int F1 = 128, F2 = 256;
    const int N  = in_sizes[0] / F1;       // 100000
    const int E  = in_sizes[1] / 2;        // 600000
    const long long* src = ei;
    const long long* dst = ei + E;

    // workspace carve-out (256B aligned)
    char* ws = (char*)d_ws;
    size_t off = 0;
    auto carve = [&](size_t bytes) -> void* {
        void* p = ws + off;
        off = (off + bytes + 255) & ~(size_t)255;
        return p;
    };
    float* deg  = (float*)carve((size_t)N * 4);
    float* dinv = (float*)carve((size_t)N * 4);
    bf16*  xb   = (bf16*) carve((size_t)N * F1 * 2);
    bf16*  W1p  = (bf16*) carve((size_t)128 * F1 * 2);
    bf16*  W2p  = (bf16*) carve((size_t)128 * F2 * 2);
    float* h1   = (float*)carve((size_t)N * F1 * 4);
    float* o1   = (float*)carve((size_t)N * F1 * 4);
    bf16*  h1b  = (bf16*) carve((size_t)N * F1 * 2);
    float* h2   = (float*)carve((size_t)N * F2 * 4);
    float* o2   = (float*)carve((size_t)N * F2 * 4);
    (void)ws_size; (void)n_in;

    // --- degrees / normalization (shared by both layers)
    k_deg_init <<<(N + 255) / 256, 256, 0, stream>>>(deg, N);
    k_deg_edges<<<(E + 255) / 256, 256, 0, stream>>>(dst, deg, E);
    k_dinv     <<<(N + 255) / 256, 256, 0, stream>>>(deg, dinv, N);

    // --- precision conversion + weight packing
    k_f2bf  <<<(N * F1 + 255) / 256, 256, 0, stream>>>(x, xb, N * F1);
    k_pack_w<<<(128 * F1 + 255) / 256, 256, 0, stream>>>(W1, W1p, 128, F1);
    k_pack_w<<<(128 * F2 + 255) / 256, 256, 0, stream>>>(W2, W2p, 128, F2);

    // --- layer 1: GEMM (WMMA) -> self-loop+bias -> edge scatter -> relu+bf16
    k_gemm_wmma<<<N / 16, 32 * (F1 / 16), 0, stream>>>(xb, W1p, h1, F1);
    k_selfloop <<<(N * F1 + 255) / 256, 256, 0, stream>>>(h1, dinv, b1, o1, 7, N * F1);
    {
        dim3 blk(F1, 4);
        k_scatter<<<(E + 3) / 4, blk, 0, stream>>>(h1, dinv, src, dst, o1, F1, E);
    }
    k_relu_f2bf<<<(N * F1 + 255) / 256, 256, 0, stream>>>(o1, h1b, N * F1);

    // --- layer 2: GEMM (WMMA) -> self-loop+bias -> edge scatter
    k_gemm_wmma<<<N / 16, 32 * (F2 / 16), 0, stream>>>(h1b, W2p, h2, F2);
    k_selfloop <<<(N * F2 + 255) / 256, 256, 0, stream>>>(h2, dinv, b2, o2, 8, N * F2);
    {
        dim3 blk(F2, 4);
        k_scatter<<<(E + 3) / 4, blk, 0, stream>>>(h2, dinv, src, dst, o2, F2, E);
    }

    // --- relu + mean over nodes -> [256]
    k_zero     <<<1, 256, 0, stream>>>(out, out_size);
    k_mean_relu<<<256, F2, 0, stream>>>(o2, out, N, F2, 1.0f / (float)N);
}
